// SplineCNN_8804682957395
// MI455X (gfx1250) — matrix operand order repaired and checked by reading, not compiled, not run on hardware
//
#include <hip/hip_runtime.h>
#include <math.h>

// ---- problem constants (fixed by the reference) ----
#define KS    5
#define KW    25        // KS*KS
#define H1C   32
#define H2C   64
#define NCOL  1600      // KW*H2C
#define NGRAPH 64
#define F1C   128
#define NOUTC 10
#define NTILES_COL (NCOL / 16)   // 100
#define GEMM_MT 10               // M-tiles per wave (B-register reuse)

typedef __attribute__((ext_vector_type(16))) _Float16 v16h;
typedef __attribute__((ext_vector_type(8)))  float    v8f;

#if defined(__has_builtin)
#if __has_builtin(__builtin_amdgcn_global_load_async_to_lds_b128) && \
    __has_builtin(__builtin_amdgcn_s_wait_asynccnt)
#define USE_ASYNC_LDS 1
#endif
#endif

#ifdef USE_ASYNC_LDS
// exact parameter types per hipcc diagnostic: vector_size(16) int, as1 / as3
typedef int v4i_vs __attribute__((vector_size(16)));
typedef __attribute__((address_space(1))) v4i_vs* g128p;
typedef __attribute__((address_space(3))) v4i_vs* l128p;
#endif

// wave-uniform value -> SGPR (compiler can't prove threadIdx.x>>5 is uniform)
__device__ __forceinline__ int wave_uniform(int v) {
  return __builtin_amdgcn_readfirstlane(v);
}

__device__ __forceinline__ void spline_basis(const float* __restrict__ attr, int e,
                                             int& k0, int& k1, float& f0, float& f1) {
  float v0 = attr[2 * e]     * (float)(KS - 1);
  float v1 = attr[2 * e + 1] * (float)(KS - 1);
  int a0 = (int)floorf(v0); a0 = a0 < 0 ? 0 : (a0 > KS - 2 ? KS - 2 : a0);
  int a1 = (int)floorf(v1); a1 = a1 < 0 ? 0 : (a1 > KS - 2 ? KS - 2 : a1);
  k0 = a0; k1 = a1;
  f0 = v0 - (float)a0;
  f1 = v1 - (float)a1;
}

// ---------------- layer 1: edge messages + degree (atomics into L2) ----------
__global__ __launch_bounds__(256) void edge1_kernel(
    const float* __restrict__ x, const float* __restrict__ attr,
    const int* __restrict__ eidx, const float* __restrict__ W1,
    float* __restrict__ agg1, float* __restrict__ deg, int N, int E) {
  __shared__ float sW1[KW * H1C];
  for (int i = threadIdx.x; i < KW * H1C; i += blockDim.x) sW1[i] = W1[i];
  __syncthreads();
  int e = blockIdx.x * 8 + wave_uniform(threadIdx.x >> 5);  // scalar edge id
  int c = threadIdx.x & 31;
  if (e >= E) return;                                       // scalar branch
  int src = eidx[e], dst = eidx[E + e];                     // uniform loads
  int k0, k1; float f0, f1;
  spline_basis(attr, e, k0, k1, f0, f1);
  float w00 = (1.f - f0) * (1.f - f1), w10 = f0 * (1.f - f1);
  float w01 = (1.f - f0) * f1,          w11 = f0 * f1;
  int i00 = (k0 + k1 * KS) * H1C;  // wi = (k0+b0) + (k1+b1)*KS
  float ws = w00 * sW1[i00 + c]
           + w10 * sW1[i00 + H1C + c]
           + w01 * sW1[i00 + KS * H1C + c]
           + w11 * sW1[i00 + (KS + 1) * H1C + c];
  float msg = x[src] * ws;
  atomicAdd(&agg1[(size_t)dst * H1C + c], msg);
  if (c == 0) atomicAdd(&deg[dst], 1.0f);
}

// ---------------- layer 1: node update (mean + root + bias, ReLU) ------------
__global__ void node1_kernel(const float* __restrict__ x, const float* __restrict__ agg1,
                             const float* __restrict__ deg, const float* __restrict__ root1,
                             const float* __restrict__ b1, float* __restrict__ h1,
                             _Float16* __restrict__ Ah, int N) {
  int t = blockIdx.x * blockDim.x + threadIdx.x;
  if (t >= N * H1C) return;
  int n = t >> 5, c = t & 31;
  float d = fmaxf(deg[n], 1.0f);
  float v = agg1[t] / d + x[n] * root1[c] + b1[c];
  v = fmaxf(v, 0.0f);
  h1[t] = v;
  Ah[t] = (_Float16)v;   // f16 copy for WMMA A-matrix
}

// ---------------- reshape W2 [25,32,64] -> B f16 [32,1600] -------------------
__global__ void prep_b_kernel(const float* __restrict__ W2, _Float16* __restrict__ Bh) {
  int idx = blockIdx.x * blockDim.x + threadIdx.x;
  if (idx >= H1C * NCOL) return;
  int ci = idx / NCOL, rest = idx % NCOL;
  int wi = rest / H2C, co = rest % H2C;
  Bh[idx] = (_Float16)W2[(size_t)wi * (H1C * H2C) + ci * H2C + co];
}

// ---------------- WMMA GEMM: xW2h[N,1600] = Ah[N,32] x Bh[32,1600] -----------
// Each wave owns one 16-wide column tile (B kept in registers) and sweeps
// GEMM_MT row tiles; one v_wmma_f32_16x16x32_f16 per tile (K=32 = one step).
// A tiles (1KB each) are double-buffered through LDS with gfx1250 async copies.
__global__ __launch_bounds__(256) void gemm_xw2_kernel(
    const _Float16* __restrict__ Ah, const _Float16* __restrict__ Bh,
    _Float16* __restrict__ Ch, int N, int totalWaves) {
#ifdef USE_ASYNC_LDS
  __shared__ _Float16 sA[8][2][16 * H1C];   // 8 waves x 2 buffers x 1KB
#endif
  int wid  = wave_uniform(threadIdx.x >> 5);               // scalar wave-in-block
  int wave = blockIdx.x * 8 + wid;                         // scalar global wave
  int lane = threadIdx.x & 31;
  if (wave >= totalWaves) return;                          // scalar branch
  int tn  = wave % NTILES_COL;                             // scalar
  int tm0 = (wave / NTILES_COL) * GEMM_MT;                 // scalar
  if (tm0 * 16 >= N) return;

  // B-matrix 32x16 f16: lane&15 = N column, 16 halves = K in table order
  v16h bm;
#pragma unroll
  for (int s = 0; s < 16; ++s) {
    int kk = ((lane >> 4) << 3) + ((s >> 3) << 4) + (s & 7);
    bm[s] = Bh[(size_t)kk * NCOL + tn * 16 + (lane & 15)];
  }
  int col = tn * 16 + (lane & 15);
  int mloc = lane & 15;

#ifdef USE_ASYNC_LDS
  // kick off the first full tile copy into buffer 0 (2 x b128 per lane = 1KB)
  if ((tm0 + 1) * 16 <= N) {                               // scalar
    g128p g = (g128p)((const char*)(Ah + (size_t)tm0 * 16 * H1C) + lane * 32);
    l128p l = (l128p)((char*)&sA[wid][0][0] + lane * 32);
    __builtin_amdgcn_global_load_async_to_lds_b128(g, l, 0, 0);
    __builtin_amdgcn_global_load_async_to_lds_b128(g, l, 16, 0);  // imm offset hits both sides
  }
#endif

#pragma unroll 1
  for (int i = 0; i < GEMM_MT; ++i) {
    int tm = tm0 + i;                                      // scalar
    if (tm * 16 >= N) break;                               // scalar
    bool full = ((tm + 1) * 16 <= N);                      // scalar

    v16h a = {};
#ifdef USE_ASYNC_LDS
    if (full) {
      __builtin_amdgcn_s_wait_asynccnt(0);
      // prefetch next tile into the other buffer while we compute this one
      int tmn = tm + 1;
      if (i + 1 < GEMM_MT && (tmn + 1) * 16 <= N) {        // scalar
        g128p g = (g128p)((const char*)(Ah + (size_t)tmn * 16 * H1C) + lane * 32);
        l128p l = (l128p)((char*)&sA[wid][(i + 1) & 1][0] + lane * 32);
        __builtin_amdgcn_global_load_async_to_lds_b128(g, l, 0, 0);
        __builtin_amdgcn_global_load_async_to_lds_b128(g, l, 16, 0);
      }
      const _Float16* at = &sA[wid][i & 1][0];
#pragma unroll
      for (int p = 0; p < 8; ++p) {
        int kb = ((lane >> 4) << 3) + ((p >> 2) << 4) + ((p & 3) << 1);
        a[2 * p]     = at[mloc * H1C + kb];
        a[2 * p + 1] = at[mloc * H1C + kb + 1];
      }
    } else
#endif
    {
      // direct global loads (fallback / partial boundary tile)
      int m = tm * 16 + mloc;
      if (m < N) {
        const _Float16* ar = Ah + (size_t)m * H1C;
#pragma unroll
        for (int p = 0; p < 8; ++p) {
          int kb = ((lane >> 4) << 3) + ((p >> 2) << 4) + ((p & 3) << 1);
          union { unsigned u; _Float16 h[2]; } t;
          t.u = *(const unsigned*)(ar + kb);
          a[2 * p]     = t.h[0];
          a[2 * p + 1] = t.h[1];
        }
      }
    }

    v8f c = {};
    c = __builtin_amdgcn_wmma_f32_16x16x32_f16(
        /*neg_a=*/false, a, /*neg_b=*/false, bm,
        /*c_mod=*/(short)0, c, /*reuse_a=*/false, /*reuse_b=*/false);

    // D layout: VGPR r -> M = r + (lane>=16 ? 8 : 0), N = lane&15
    int mbase = tm * 16 + ((lane >= 16) ? 8 : 0);
    _Float16* cp = Ch + (size_t)mbase * NCOL + col;
    if (full) {                              // scalar branch: straight-line stores
#pragma unroll
      for (int r = 0; r < 8; ++r) cp[(size_t)r * NCOL] = (_Float16)c[r];
    } else {
#pragma unroll
      for (int r = 0; r < 8; ++r)
        if (mbase + r < N) cp[(size_t)r * NCOL] = (_Float16)c[r];
    }
  }
}

// ---------------- layer 2: edge gather (L2-resident xW2h) + scatter ----------
// One wave per edge; each lane handles 2 channels via dword loads.
__global__ __launch_bounds__(256) void edge2_kernel(
    const float* __restrict__ attr, const int* __restrict__ eidx,
    const _Float16* __restrict__ xW2h, float* __restrict__ agg2, int E) {
  int e = blockIdx.x * 8 + wave_uniform(threadIdx.x >> 5);  // scalar edge id
  int l = threadIdx.x & 31;                 // channels 2l, 2l+1
  if (e >= E) return;                       // scalar branch
  int src = eidx[e], dst = eidx[E + e];     // uniform loads
  int k0, k1; float f0, f1;
  spline_basis(attr, e, k0, k1, f0, f1);
  float w00 = (1.f - f0) * (1.f - f1), w10 = f0 * (1.f - f1);
  float w01 = (1.f - f0) * f1,          w11 = f0 * f1;
  const unsigned* base =
      (const unsigned*)(xW2h + (size_t)src * NCOL + (k0 + k1 * KS) * H2C);
  union { unsigned u; _Float16 h[2]; } t0, t1, t2, t3;
  t0.u = base[l];                          // (k0,   k1  )
  t1.u = base[(H2C / 2) + l];              // (k0+1, k1  )
  t2.u = base[KS * (H2C / 2) + l];         // (k0,   k1+1)
  t3.u = base[(KS + 1) * (H2C / 2) + l];   // (k0+1, k1+1)
  float m0 = w00 * (float)t0.h[0] + w10 * (float)t1.h[0]
           + w01 * (float)t2.h[0] + w11 * (float)t3.h[0];
  float m1 = w00 * (float)t0.h[1] + w10 * (float)t1.h[1]
           + w01 * (float)t2.h[1] + w11 * (float)t3.h[1];
  float* dp = agg2 + (size_t)dst * H2C + 2 * l;
  atomicAdd(dp,     m0);
  atomicAdd(dp + 1, m1);
}

// ---------------- layer 2: node update ---------------------------------------
__global__ void node2_kernel(const float* __restrict__ agg2, const float* __restrict__ deg,
                             const float* __restrict__ h1, const float* __restrict__ root2,
                             const float* __restrict__ b2, float* __restrict__ h2, int N) {
  int t = blockIdx.x * blockDim.x + threadIdx.x;
  if (t >= N * H2C) return;
  int n = t >> 6, c = t & 63;
  float d = fmaxf(deg[n], 1.0f);
  float acc = agg2[t] / d + b2[c];
  const float* hr = h1 + (size_t)n * H1C;
#pragma unroll
  for (int ci = 0; ci < H1C; ++ci) acc += hr[ci] * root2[ci * H2C + c];
  h2[t] = fmaxf(acc, 0.0f);
}

// ---------------- global mean pool -------------------------------------------
__global__ void pool_kernel(const float* __restrict__ h2, const int* __restrict__ batch,
                            float* __restrict__ gsum, float* __restrict__ cnt, int N) {
  int t = blockIdx.x * blockDim.x + threadIdx.x;
  if (t >= N * H2C) return;
  int n = t >> 6, c = t & 63;
  int b = batch[n];
  atomicAdd(&gsum[b * H2C + c], h2[t]);
  if (c == 0) atomicAdd(&cnt[b], 1.0f);
}

// ---------------- FC1 (64 -> 128) + ReLU -------------------------------------
__global__ __launch_bounds__(F1C) void fc1_kernel(
    const float* __restrict__ gsum, const float* __restrict__ cnt,
    const float* __restrict__ Wf1, const float* __restrict__ bf1,
    float* __restrict__ gact) {
  __shared__ float gs[H2C];
  int b = blockIdx.x;
  if (threadIdx.x < H2C)
    gs[threadIdx.x] = gsum[b * H2C + threadIdx.x] / fmaxf(cnt[b], 1.0f);
  __syncthreads();
  int j = threadIdx.x;
  float acc = bf1[j];
#pragma unroll
  for (int c = 0; c < H2C; ++c) acc += gs[c] * Wf1[c * F1C + j];
  gact[b * F1C + j] = fmaxf(acc, 0.0f);
}

// ---------------- FC2 (128 -> 10) + log_softmax ------------------------------
__global__ void out_kernel(const float* __restrict__ gact, const float* __restrict__ Wf2,
                           const float* __restrict__ bf2, float* __restrict__ out) {
  int b = blockIdx.x * blockDim.x + threadIdx.x;
  if (b >= NGRAPH) return;
  const float* g = gact + b * F1C;
  float l[NOUTC];
#pragma unroll
  for (int o = 0; o < NOUTC; ++o) {
    float acc = bf2[o];
    for (int c = 0; c < F1C; ++c) acc += g[c] * Wf2[c * NOUTC + o];
    l[o] = acc;
  }
  float m = l[0];
#pragma unroll
  for (int o = 1; o < NOUTC; ++o) m = fmaxf(m, l[o]);
  float s = 0.f;
#pragma unroll
  for (int o = 0; o < NOUTC; ++o) s += expf(l[o] - m);
  float lse = m + logf(s);
#pragma unroll
  for (int o = 0; o < NOUTC; ++o) out[b * NOUTC + o] = l[o] - lse;
}

// =============================================================================
extern "C" void kernel_launch(void* const* d_in, const int* in_sizes, int n_in,
                              void* d_out, int out_size, void* d_ws, size_t ws_size,
                              hipStream_t stream) {
  const float* x     = (const float*)d_in[0];
  const float* attr  = (const float*)d_in[1];
  const float* W1    = (const float*)d_in[2];
  const float* root1 = (const float*)d_in[3];
  const float* b1    = (const float*)d_in[4];
  const float* W2    = (const float*)d_in[5];
  const float* root2 = (const float*)d_in[6];
  const float* b2    = (const float*)d_in[7];
  const float* Wf1   = (const float*)d_in[8];
  const float* bf1   = (const float*)d_in[9];
  const float* Wf2   = (const float*)d_in[10];
  const float* bf2   = (const float*)d_in[11];
  const int*   eidx  = (const int*)d_in[12];
  const int*   batch = (const int*)d_in[13];
  float* out = (float*)d_out;

  const int N = in_sizes[0];        // x is [N,1]
  const int E = in_sizes[1] / 2;    // edge_attr is [E,2]

  // ---- carve workspace (256B aligned slices) ----
  char* w = (char*)d_ws;
  auto carve = [&](size_t bytes) -> void* {
    void* p = (void*)w;
    w += (bytes + 255) & ~((size_t)255);
    return p;
  };
  float*     deg   = (float*)carve((size_t)N * sizeof(float));
  float*     agg1  = (float*)carve((size_t)N * H1C * sizeof(float));
  float*     h1    = (float*)carve((size_t)N * H1C * sizeof(float));
  _Float16*  Ah    = (_Float16*)carve((size_t)N * H1C * sizeof(_Float16));
  _Float16*  Bh    = (_Float16*)carve((size_t)H1C * NCOL * sizeof(_Float16));
  _Float16*  xW2h  = (_Float16*)carve((size_t)N * NCOL * sizeof(_Float16));
  float*     agg2  = (float*)carve((size_t)N * H2C * sizeof(float));
  float*     h2    = (float*)carve((size_t)N * H2C * sizeof(float));
  float*     gsum  = (float*)carve((size_t)NGRAPH * H2C * sizeof(float));
  float*     cnt   = (float*)carve((size_t)NGRAPH * sizeof(float));
  float*     gact  = (float*)carve((size_t)NGRAPH * F1C * sizeof(float));

  // ---- zero accumulation buffers (graph-capture-legal async memsets) ----
  (void)hipMemsetAsync(deg,  0, (size_t)N * sizeof(float), stream);
  (void)hipMemsetAsync(agg1, 0, (size_t)N * H1C * sizeof(float), stream);
  (void)hipMemsetAsync(agg2, 0, (size_t)N * H2C * sizeof(float), stream);
  (void)hipMemsetAsync(gsum, 0, (size_t)NGRAPH * H2C * sizeof(float), stream);
  (void)hipMemsetAsync(cnt,  0, (size_t)NGRAPH * sizeof(float), stream);

  // ---- layer 1 ----
  edge1_kernel<<<(E + 7) / 8, 256, 0, stream>>>(x, attr, eidx, W1, agg1, deg, N, E);
  node1_kernel<<<((size_t)N * H1C + 255) / 256, 256, 0, stream>>>(x, agg1, deg, root1, b1,
                                                                  h1, Ah, N);

  // ---- layer 2: WMMA GEMM then gather/scatter ----
  prep_b_kernel<<<(H1C * NCOL + 255) / 256, 256, 0, stream>>>(W2, Bh);
  int mtiles = (N + 15) / 16;
  int totalWaves = ((mtiles + GEMM_MT - 1) / GEMM_MT) * NTILES_COL;
  gemm_xw2_kernel<<<(totalWaves + 7) / 8, 256, 0, stream>>>(Ah, Bh, xW2h, N, totalWaves);
  edge2_kernel<<<(E + 7) / 8, 256, 0, stream>>>(attr, eidx, xW2h, agg2, E);
  node2_kernel<<<((size_t)N * H2C + 255) / 256, 256, 0, stream>>>(agg2, deg, h1, root2, b2,
                                                                  h2, N);

  // ---- pool + MLP head ----
  pool_kernel<<<((size_t)N * H2C + 255) / 256, 256, 0, stream>>>(h2, batch, gsum, cnt, N);
  fc1_kernel<<<NGRAPH, F1C, 0, stream>>>(gsum, cnt, Wf1, bf1, gact);
  out_kernel<<<1, NGRAPH, 0, stream>>>(gact, Wf2, bf2, out);
}
